// Net_21646635172365
// MI455X (gfx1250) — compile-verified, matrix operand order checked by reading, not run on hardware
//
#include <hip/hip_runtime.h>
#include <hip/hip_bf16.h>

// GAT 2-layer forward for MI455X (gfx1250, wave32, WMMA).
// Layer1: h1 = X@W1 (bf16 WMMA, f32 acc) -> per-head attention softmax over
// incoming edges -> aggregate -> +bias -> ELU. Layer2 same with C=16, heads=1,
// then log_softmax.

#define FDIM 512
#define HD   64     // H*D layer-1 width
#define CCLS 16

typedef __attribute__((ext_vector_type(16))) __bf16        v16bf;
typedef __attribute__((ext_vector_type(8)))  float          v8f;

// order-preserving float<->uint encoding for atomicMax on floats
static __device__ __forceinline__ unsigned enc_f32(float f) {
  unsigned u = __float_as_uint(f);
  return (u & 0x80000000u) ? ~u : (u | 0x80000000u);
}
static __device__ __forceinline__ float dec_f32(unsigned u) {
  unsigned b = (u & 0x80000000u) ? (u & 0x7FFFFFFFu) : ~u;
  return __uint_as_float(b);
}

__global__ void fill_u32(unsigned* __restrict__ p, unsigned v, long long n) {
  long long i = (long long)blockIdx.x * blockDim.x + threadIdx.x;
  long long st = (long long)gridDim.x * blockDim.x;
  for (; i < n; i += st) p[i] = v;
}

// ---------------------------------------------------------------------------
// Dense GEMM Hout[nRows,NCOLS] = A[nRows,KDIM] @ W[KDIM,NCOLS] via
// v_wmma_f32_16x16x32_bf16. 8 waves/block, 16 rows/wave (128 rows/block).
// W panel staged in LDS pre-swizzled into B-fragment order; native f32->bf16
// converts (v_cvt_pk_bf16_f32 class) instead of manual bit twiddling.
// ---------------------------------------------------------------------------
template <int KDIM, int NCOLS>
__global__ __launch_bounds__(256) void gat_wmma_gemm(
    const float* __restrict__ A, const float* __restrict__ W,
    float* __restrict__ Hout, int nRows) {
  constexpr int KSTEPS = KDIM / 32;
  constexpr int NTILES = NCOLS / 16;
  __shared__ __bf16 ldsW[KSTEPS * NTILES * 512];

  const int tid = threadIdx.x;
  // Stage W as bf16 in B-fragment order: frag=(ks,nt), within frag lane*16+j.
  // Lane li supplies column n = nt*16 + (li&15); element j -> K =
  // ks*32 + (li<16?0:8) + j + (j>=8?8:0)   (CDNA5 16-bit operand layout).
  for (int i = tid; i < KSTEPS * NTILES * 512; i += 256) {
    int frag = i >> 9;
    int li   = (i >> 4) & 31;
    int j    = i & 15;
    int ks   = frag / NTILES;
    int nt   = frag % NTILES;
    int kb   = (li < 16) ? 0 : 8;
    int k    = ks * 32 + kb + j + ((j >= 8) ? 8 : 0);
    int n    = nt * 16 + (li & 15);
    ldsW[i]  = (__bf16)W[k * NCOLS + n];
  }
  __syncthreads();

  const int wave     = tid >> 5;
  const int lane     = tid & 31;
  const int rowBase  = blockIdx.x * 128 + wave * 16;
  const int laneHalf = (lane >= 16) ? 8 : 0;
  int m = rowBase + (lane & 15);
  if (m >= nRows) m = nRows - 1;   // clamp loads; stores are masked below

  v8f acc[NTILES] = {};

  const int kb = (lane < 16) ? 0 : 8;
  for (int ks = 0; ks < KSTEPS; ++ks) {
    const float* ap = A + (long long)m * KDIM + ks * 32 + kb;
    float4 x0 = *(const float4*)(ap + 0);
    float4 x1 = *(const float4*)(ap + 4);
    float4 x2 = *(const float4*)(ap + 16);
    float4 x3 = *(const float4*)(ap + 20);
    if (ks + 1 < KSTEPS) __builtin_prefetch(ap + 32, 0, 1);  // global_prefetch_b8

    v16bf a;
    a[0]  = (__bf16)x0.x; a[1]  = (__bf16)x0.y;
    a[2]  = (__bf16)x0.z; a[3]  = (__bf16)x0.w;
    a[4]  = (__bf16)x1.x; a[5]  = (__bf16)x1.y;
    a[6]  = (__bf16)x1.z; a[7]  = (__bf16)x1.w;
    a[8]  = (__bf16)x2.x; a[9]  = (__bf16)x2.y;
    a[10] = (__bf16)x2.z; a[11] = (__bf16)x2.w;
    a[12] = (__bf16)x3.x; a[13] = (__bf16)x3.y;
    a[14] = (__bf16)x3.z; a[15] = (__bf16)x3.w;

#pragma unroll
    for (int nt = 0; nt < NTILES; ++nt) {
      v16bf b = *(const v16bf*)&ldsW[(ks * NTILES + nt) * 512 + lane * 16];
      acc[nt] = __builtin_amdgcn_wmma_f32_16x16x32_bf16(
          false, a, false, b, (short)0, acc[nt], false, false);
    }
  }

  // C/D layout: VGPR v holds M = v (lanes 0-15) / v+8 (lanes 16-31), N = lane&15
#pragma unroll
  for (int nt = 0; nt < NTILES; ++nt) {
#pragma unroll
    for (int v = 0; v < 8; ++v) {
      int mr = rowBase + laneHalf + v;
      if (mr < nRows)
        Hout[(long long)mr * NCOLS + nt * 16 + (lane & 15)] = acc[nt][v];
    }
  }
}

// ---------------------------------------------------------------------------
// Attention coefficients
// ---------------------------------------------------------------------------
__global__ void gat_alpha1(const float* __restrict__ h1,
                           const float* __restrict__ a_s,
                           const float* __restrict__ a_d,
                           float* __restrict__ as, float* __restrict__ ad,
                           int Nn) {
  int t = blockIdx.x * blockDim.x + threadIdx.x;
  if (t >= Nn * 8) return;
  int n = t >> 3, h = t & 7;
  const float* hp = h1 + (long long)n * HD + h * 8;
  float s = 0.f, d = 0.f;
#pragma unroll
  for (int i = 0; i < 8; ++i) {
    s += hp[i] * a_s[h * 8 + i];
    d += hp[i] * a_d[h * 8 + i];
  }
  as[t] = s; ad[t] = d;
}

__global__ void gat_alpha2(const float* __restrict__ h2,
                           const float* __restrict__ a_s,
                           const float* __restrict__ a_d,
                           float* __restrict__ as, float* __restrict__ ad,
                           int Nn) {
  int n = blockIdx.x * blockDim.x + threadIdx.x;
  if (n >= Nn) return;
  float s = 0.f, d = 0.f;
#pragma unroll
  for (int c = 0; c < CCLS; ++c) {
    float v = h2[(long long)n * CCLS + c];
    s += v * a_s[c];
    d += v * a_d[c];
  }
  as[n] = s; ad[n] = d;
}

// ---------------------------------------------------------------------------
// Edge passes (Hh = heads: 8 for layer1, 1 for layer2)
// ---------------------------------------------------------------------------
__global__ void gat_edge_max(const int* __restrict__ src, const int* __restrict__ dst,
                             const float* __restrict__ as, const float* __restrict__ ad,
                             unsigned* __restrict__ mx, int E0, int ET, int Hh) {
  long long t = (long long)blockIdx.x * blockDim.x + threadIdx.x;
  if (t >= (long long)ET * Hh) return;
  int e = (int)(t / Hh), h = (int)(t % Hh);
  int s = (e < E0) ? src[e] : (e - E0);
  int d = (e < E0) ? dst[e] : (e - E0);
  float v = as[(long long)s * Hh + h] + ad[(long long)d * Hh + h];
  v = (v > 0.f) ? v : 0.2f * v;
  atomicMax(&mx[(long long)d * Hh + h], enc_f32(v));
}

__global__ void gat_edge_sum(const int* __restrict__ src, const int* __restrict__ dst,
                             const float* __restrict__ as, const float* __restrict__ ad,
                             const unsigned* __restrict__ mx, float* __restrict__ den,
                             int E0, int ET, int Hh) {
  long long t = (long long)blockIdx.x * blockDim.x + threadIdx.x;
  if (t >= (long long)ET * Hh) return;
  int e = (int)(t / Hh), h = (int)(t % Hh);
  int s = (e < E0) ? src[e] : (e - E0);
  int d = (e < E0) ? dst[e] : (e - E0);
  float v = as[(long long)s * Hh + h] + ad[(long long)d * Hh + h];
  v = (v > 0.f) ? v : 0.2f * v;
  atomicAdd(&den[(long long)d * Hh + h], expf(v - dec_f32(mx[(long long)d * Hh + h])));
}

// layer1 aggregate: 16 lanes per edge, 4 channels/lane (64 ch, head = ch/8)
__global__ void gat_edge_agg1(const int* __restrict__ src, const int* __restrict__ dst,
                              const float* __restrict__ as, const float* __restrict__ ad,
                              const unsigned* __restrict__ mx, const float* __restrict__ den,
                              const float* __restrict__ h1, float* __restrict__ agg,
                              int E0, int ET) {
  long long t = (long long)blockIdx.x * blockDim.x + threadIdx.x;
  long long g = t >> 4;
  int l = (int)(t & 15);
  if (g >= ET) return;
  int e = (int)g;
  int s = (e < E0) ? src[e] : (e - E0);
  int d = (e < E0) ? dst[e] : (e - E0);
  int head = l >> 1;  // channels l*4..l*4+3 all belong to head l/2
  float v = as[(long long)s * 8 + head] + ad[(long long)d * 8 + head];
  v = (v > 0.f) ? v : 0.2f * v;
  long long dh = (long long)d * 8 + head;
  float w = expf(v - dec_f32(mx[dh])) / (den[dh] + 1e-16f);
  int c0 = l * 4;
  const float4 hv = *(const float4*)(h1 + (long long)s * HD + c0);
  float* o = agg + (long long)d * HD + c0;
  atomicAdd(o + 0, w * hv.x);
  atomicAdd(o + 1, w * hv.y);
  atomicAdd(o + 2, w * hv.z);
  atomicAdd(o + 3, w * hv.w);
}

// layer2 aggregate: 16 lanes per edge, 1 channel/lane
__global__ void gat_edge_agg2(const int* __restrict__ src, const int* __restrict__ dst,
                              const float* __restrict__ as, const float* __restrict__ ad,
                              const unsigned* __restrict__ mx, const float* __restrict__ den,
                              const float* __restrict__ h2, float* __restrict__ agg,
                              int E0, int ET) {
  long long t = (long long)blockIdx.x * blockDim.x + threadIdx.x;
  long long g = t >> 4;
  int l = (int)(t & 15);
  if (g >= ET) return;
  int e = (int)g;
  int s = (e < E0) ? src[e] : (e - E0);
  int d = (e < E0) ? dst[e] : (e - E0);
  float v = as[s] + ad[d];
  v = (v > 0.f) ? v : 0.2f * v;
  float w = expf(v - dec_f32(mx[d])) / (den[d] + 1e-16f);
  atomicAdd(&agg[(long long)d * CCLS + l], w * h2[(long long)s * CCLS + l]);
}

// ---------------------------------------------------------------------------
// Finalizers
// ---------------------------------------------------------------------------
__global__ void gat_fin1(float* __restrict__ agg, const float* __restrict__ bias,
                         long long n64) {  // in place: +bias then ELU
  long long t = (long long)blockIdx.x * blockDim.x + threadIdx.x;
  if (t >= n64) return;
  int c = (int)(t & 63);
  float x = agg[t] + bias[c];
  agg[t] = (x > 0.f) ? x : expm1f(x);
}

__global__ void gat_fin2(float* __restrict__ out, const float* __restrict__ bias,
                         int Nn) {  // in place: +bias then log_softmax over 16
  int n = blockIdx.x * blockDim.x + threadIdx.x;
  if (n >= Nn) return;
  float l[CCLS];
  float mx = -3.4e38f;
#pragma unroll
  for (int c = 0; c < CCLS; ++c) {
    l[c] = out[(long long)n * CCLS + c] + bias[c];
    mx = fmaxf(mx, l[c]);
  }
  float sum = 0.f;
#pragma unroll
  for (int c = 0; c < CCLS; ++c) sum += expf(l[c] - mx);
  float lse = mx + logf(sum);
#pragma unroll
  for (int c = 0; c < CCLS; ++c) out[(long long)n * CCLS + c] = l[c] - lse;
}

// ---------------------------------------------------------------------------
extern "C" void kernel_launch(void* const* d_in, const int* in_sizes, int n_in,
                              void* d_out, int out_size, void* d_ws, size_t ws_size,
                              hipStream_t stream) {
  (void)n_in; (void)out_size; (void)ws_size;
  const float* x        = (const float*)d_in[0];
  const int*   eidx     = (const int*)d_in[1];
  const float* W1       = (const float*)d_in[2];
  const float* att_src1 = (const float*)d_in[3];
  const float* att_dst1 = (const float*)d_in[4];
  const float* bias1    = (const float*)d_in[5];
  const float* W2       = (const float*)d_in[6];
  const float* att_src2 = (const float*)d_in[7];
  const float* att_dst2 = (const float*)d_in[8];
  const float* bias2    = (const float*)d_in[9];
  float* out = (float*)d_out;

  const int Nn = in_sizes[0] / FDIM;
  const int E0 = in_sizes[1] / 2;
  const int ET = E0 + Nn;            // + self loops
  const int* srcA = eidx;
  const int* dstA = eidx + E0;

  // workspace layout (floats)
  float*    h1   = (float*)d_ws;                 // [N,64]; reused as h2 [N,16]
  float*    agg1 = h1   + (size_t)Nn * HD;       // [N,64]; in-place -> layer2 in
  float*    as1  = agg1 + (size_t)Nn * HD;       // [N,8]
  float*    ad1  = as1  + (size_t)Nn * 8;        // [N,8]
  unsigned* mx1  = (unsigned*)(ad1 + (size_t)Nn * 8);   // [N,8]
  float*    dn1  = (float*)(mx1 + (size_t)Nn * 8);      // [N,8]
  float*    as2  = dn1 + (size_t)Nn * 8;         // [N]
  float*    ad2  = as2 + Nn;                     // [N]
  unsigned* mx2  = (unsigned*)(ad2 + Nn);        // [N]
  float*    dn2  = (float*)(mx2 + Nn);           // [N]

  const unsigned NEG_INF_ENC = 0x007FFFFFu;  // enc_f32(-inf)
  auto nb = [](long long n, int t) { return (unsigned)((n + t - 1) / t); };

  // init accumulators (every call; deterministic)
  fill_u32<<<nb((long long)Nn * 8, 256), 256, 0, stream>>>(mx1, NEG_INF_ENC, (long long)Nn * 8);
  fill_u32<<<nb((long long)Nn * 8, 256), 256, 0, stream>>>((unsigned*)dn1, 0u, (long long)Nn * 8);
  fill_u32<<<nb((long long)Nn * HD, 256), 256, 0, stream>>>((unsigned*)agg1, 0u, (long long)Nn * HD);
  fill_u32<<<nb(Nn, 256), 256, 0, stream>>>(mx2, NEG_INF_ENC, Nn);
  fill_u32<<<nb(Nn, 256), 256, 0, stream>>>((unsigned*)dn2, 0u, Nn);
  fill_u32<<<nb((long long)Nn * CCLS, 256), 256, 0, stream>>>((unsigned*)out, 0u, (long long)Nn * CCLS);

  // ---- layer 1 ----
  gat_wmma_gemm<FDIM, HD><<<nb(Nn, 128), 256, 0, stream>>>(x, W1, h1, Nn);
  gat_alpha1<<<nb((long long)Nn * 8, 256), 256, 0, stream>>>(h1, att_src1, att_dst1, as1, ad1, Nn);
  gat_edge_max<<<nb((long long)ET * 8, 256), 256, 0, stream>>>(srcA, dstA, as1, ad1, mx1, E0, ET, 8);
  gat_edge_sum<<<nb((long long)ET * 8, 256), 256, 0, stream>>>(srcA, dstA, as1, ad1, mx1, dn1, E0, ET, 8);
  gat_edge_agg1<<<nb((long long)ET * 16, 256), 256, 0, stream>>>(srcA, dstA, as1, ad1, mx1, dn1, h1, agg1, E0, ET);
  gat_fin1<<<nb((long long)Nn * HD, 256), 256, 0, stream>>>(agg1, bias1, (long long)Nn * HD);

  // ---- layer 2 ----
  gat_wmma_gemm<HD, CCLS><<<nb(Nn, 128), 256, 0, stream>>>(agg1, W2, h1 /*h2*/, Nn);
  gat_alpha2<<<nb(Nn, 256), 256, 0, stream>>>(h1, att_src2, att_dst2, as2, ad2, Nn);
  gat_edge_max<<<nb(ET, 256), 256, 0, stream>>>(srcA, dstA, as2, ad2, mx2, E0, ET, 1);
  gat_edge_sum<<<nb(ET, 256), 256, 0, stream>>>(srcA, dstA, as2, ad2, mx2, dn2, E0, ET, 1);
  gat_edge_agg2<<<nb((long long)ET * 16, 256), 256, 0, stream>>>(srcA, dstA, as2, ad2, mx2, dn2, h1, out, E0, ET);
  gat_fin2<<<nb(Nn, 256), 256, 0, stream>>>(out, bias2, Nn);
}
